// BiSRU_7834020348300
// MI455X (gfx1250) — matrix operand order repaired and checked by reading, not compiled, not run on hardware
//
#include <hip/hip_runtime.h>
#include <hip/hip_bf16.h>

// ---------------------------------------------------------------------------
// BiSRU: pre = X @ W (WMMA bf16, f32 acc, TDM-staged tiles) ->
// LayerNorm+gates (in place) -> 3-pass chunked bidirectional scan + combine.
// T=2048, B=16, D=1024 ; M = T*B = 32768, N = 3*D = 3072, K = D = 1024
// ---------------------------------------------------------------------------

#define T_DIM 2048
#define B_DIM 16
#define D_DIM 1024
#define M_DIM (T_DIM * B_DIM)      // 32768
#define N_DIM (3 * D_DIM)          // 3072
#define K_DIM D_DIM                // 1024

#define NCHUNK 16
#define CHUNK (T_DIM / NCHUNK)     // 128

typedef __bf16 v16bf __attribute__((ext_vector_type(16)));
typedef float  v8f   __attribute__((ext_vector_type(8)));
typedef unsigned int u32x4 __attribute__((ext_vector_type(4)));
typedef int          i32x4 __attribute__((ext_vector_type(4)));
typedef int          i32x8 __attribute__((ext_vector_type(8)));

union Frag { uint4 q[2]; v16bf v; };

static __device__ __forceinline__ unsigned short f32_to_bf16_rne(float f) {
    unsigned int u = __float_as_uint(f);
    u += 0x7FFFu + ((u >> 16) & 1u);           // round-to-nearest-even
    return (unsigned short)(u >> 16);
}

// ---------------------------------------------------------------------------
// Kernel 0: f32 -> bf16 conversion for X (row-major kept)
// ---------------------------------------------------------------------------
__global__ __launch_bounds__(256) void cvt_bf16_kernel(
    const float* __restrict__ src, unsigned short* __restrict__ dst, int n)
{
    int i = blockIdx.x * 256 + threadIdx.x;
    if (i < n) dst[i] = f32_to_bf16_rne(src[i]);
}

// ---------------------------------------------------------------------------
// Kernel 1: W (K,N) f32 -> Wt (N,K) bf16, LDS-tiled 32x32 transpose.
// Done once so the GEMM's B tile is a contiguous-row 2D tile (TDM-friendly,
// no ds_store_b16 scatter in the hot loop).
// ---------------------------------------------------------------------------
__global__ __launch_bounds__(256) void wt_kernel(
    const float* __restrict__ W, unsigned short* __restrict__ Wt)
{
    __shared__ unsigned short tile[32][33];
    const int n0 = blockIdx.x * 32;
    const int k0 = blockIdx.y * 32;
    const int tx = threadIdx.x & 31;
    const int ty = threadIdx.x >> 5;           // 8 rows per pass
#pragma unroll
    for (int i = 0; i < 32; i += 8)
        tile[ty + i][tx] = f32_to_bf16_rne(W[(size_t)(k0 + ty + i) * N_DIM + n0 + tx]);
    __syncthreads();
#pragma unroll
    for (int i = 0; i < 32; i += 8)
        Wt[(size_t)(n0 + ty + i) * K_DIM + k0 + tx] = tile[tx][ty + i];
}

// ---------------------------------------------------------------------------
// Kernel 2: GEMM pre[M,N] = Xb[M,K] * Wt[N,K]^T   (bf16 in, f32 out)
// Block: 256 thr = 8 waves; tile 128(M) x 128(N), BK=64.
// Waves 4(M) x 2(N), each 32x64 = 8 WMMA tiles, 16 WMMA per K iter.
// Tiles staged by the Tensor Data Mover (double-buffered, TENSORcnt ping-pong)
// when the builtin exists; manual b128 staging otherwise.
// LDS row stride = 72 shorts (144 B) -> 16 fragment rows hit 16 distinct banks.
// ---------------------------------------------------------------------------
#define BK 64
#define LDP 72            // shorts per LDS row (64 data + 8 pad)
#define NKT (K_DIM / BK)  // 16

#if __has_builtin(__builtin_amdgcn_tensor_load_to_lds) && \
    __has_builtin(__builtin_amdgcn_s_wait_tensorcnt)
#define USE_TDM 1
#endif

#ifdef USE_TDM
// Build D# (groups 0/1) for a 2D bf16 tile: tile_dim0=64 (k), tile_dim1=128
// (rows), tensor_dim0=1024 (=K row length), dim0_stride=1024, padded so each
// 128B LDS row gets +16B (pad_interval code 4 = 32 DWORDs, pad_amount code 3
// = 4 DWORDs) -> LDS row stride 144 B.
static __device__ __forceinline__ void tdm_load_tile_2d(
    const unsigned short* gbase, size_t elem_off, int rows_dim1, unsigned lds_off)
{
    unsigned long long ga = (unsigned long long)(uintptr_t)gbase
                          + ((unsigned long long)elem_off << 1);
    u32x4 g0;
    g0.x = 1u;                                            // count=1
    g0.y = lds_off;                                       // LDS byte address
    g0.z = (unsigned)(ga & 0xFFFFFFFFull);                // global_addr[31:0]
    g0.w = (unsigned)((ga >> 32) & 0x1FFFFFFu) | (2u << 30); // [56:32] | type=2
    i32x8 g1;
    g1.s0 = (int)((1u << 16) | (1u << 20) | (4u << 22) | (3u << 25));
    g1.s1 = (int)(1024u << 16);                 // tensor_dim0 = 1024 (lo16)
    g1.s2 = (int)((unsigned)rows_dim1 << 16);   // dim0 hi=0 | tensor_dim1 lo16
    g1.s3 = (int)(64u << 16);                   // dim1 hi=0 | tile_dim0 = 64
    g1.s4 = 128;                                // tile_dim1=128, tile_dim2=0
    g1.s5 = 1024;                               // tensor_dim0_stride lo32
    g1.s6 = 0;
    g1.s7 = 0;
    i32x4 z4 = {0, 0, 0, 0};
#if __has_include(<hip/amd_detail/amd_gfx1250_TDM.h>)
    i32x8 z8 = {0, 0, 0, 0, 0, 0, 0, 0};
    __builtin_amdgcn_tensor_load_to_lds(g0, g1, z4, z4, z8, 0);   // 6-arg lane
#else
    __builtin_amdgcn_tensor_load_to_lds(g0, g1, z4, z4, 0);       // 5-arg lane
#endif
}
#endif

__global__ __launch_bounds__(256) void gemm_bf16_kernel(
    const unsigned short* __restrict__ Xb,   // [M,K] bf16 bits
    const unsigned short* __restrict__ Wt,   // [N,K] bf16 bits
    float* __restrict__ pre)                 // [M,N] f32
{
    __shared__ __align__(16) unsigned short As[2][128 * LDP];
    __shared__ __align__(16) unsigned short Bs[2][128 * LDP];

    const int m0 = blockIdx.y * 128;
    const int n0 = blockIdx.x * 128;

    const int tid   = threadIdx.x;
    const int wave  = tid >> 5;
    const int lane  = tid & 31;
    const int lmod  = lane & 15;
    const int lhalf = lane >> 4;

    const int wm = (wave & 3) * 32;
    const int wn = (wave >> 2) * 64;

    v8f acc[2][4];
#pragma unroll
    for (int mi = 0; mi < 2; ++mi)
#pragma unroll
        for (int ni = 0; ni < 4; ++ni)
            acc[mi][ni] = (v8f)0.0f;

#ifdef USE_TDM
    if (tid < 32) {   // wave 0 drives the DMA; descriptors are uniform (SGPR)
        tdm_load_tile_2d(Xb, (size_t)m0 * K_DIM + 0, M_DIM,
                         (unsigned)(uintptr_t)&As[0][0]);
        tdm_load_tile_2d(Wt, (size_t)n0 * K_DIM + 0, N_DIM,
                         (unsigned)(uintptr_t)&Bs[0][0]);
    }
#endif

    for (int kt = 0; kt < NKT; ++kt) {
        const int buf = kt & 1;
#ifdef USE_TDM
        if (tid < 32) {
            if (kt + 1 < NKT) {
                tdm_load_tile_2d(Xb, (size_t)m0 * K_DIM + (kt + 1) * BK, M_DIM,
                                 (unsigned)(uintptr_t)&As[buf ^ 1][0]);
                tdm_load_tile_2d(Wt, (size_t)n0 * K_DIM + (kt + 1) * BK, N_DIM,
                                 (unsigned)(uintptr_t)&Bs[buf ^ 1][0]);
                __builtin_amdgcn_s_wait_tensorcnt((short)2);  // current pair done
            } else {
                __builtin_amdgcn_s_wait_tensorcnt((short)0);
            }
        }
        __syncthreads();
#else
        {   // manual staging (coalesced b128 both sides), single buffer
            const int row = tid >> 1;
            const int seg = (tid & 1) * 32;                // shorts
            const uint4* sa = (const uint4*)(Xb + (size_t)(m0 + row) * K_DIM + kt * BK + seg);
            const uint4* sb = (const uint4*)(Wt + (size_t)(n0 + row) * K_DIM + kt * BK + seg);
#pragma unroll
            for (int j = 0; j < 4; ++j) {
                *(uint4*)&As[0][row * LDP + seg + j * 8] = sa[j];
                *(uint4*)&Bs[0][row * LDP + seg + j * 8] = sb[j];
            }
        }
        __syncthreads();
#endif
        const int cbuf = (
#ifdef USE_TDM
            buf
#else
            0
#endif
        );
        // two 32-wide K steps per BK=64 tile
#pragma unroll
        for (int s = 0; s < 2; ++s) {
            Frag afr[2], bfr[4];
#pragma unroll
            for (int mi = 0; mi < 2; ++mi) {
                const int r = wm + mi * 16 + lmod;
                const uint4* p = (const uint4*)&As[cbuf][r * LDP + s * 32];
                afr[mi].q[0] = p[lhalf];         // K = 8*half..+7
                afr[mi].q[1] = p[2 + lhalf];     // K = 16+8*half..+7
            }
#pragma unroll
            for (int ni = 0; ni < 4; ++ni) {
                const int c = wn + ni * 16 + lmod;
                const uint4* p = (const uint4*)&Bs[cbuf][c * LDP + s * 32];
                bfr[ni].q[0] = p[lhalf];         // K = 8*half..+7
                bfr[ni].q[1] = p[2 + lhalf];     // K = 16+8*half..+7
            }
#pragma unroll
            for (int mi = 0; mi < 2; ++mi)
#pragma unroll
                for (int ni = 0; ni < 4; ++ni)
                    acc[mi][ni] = __builtin_amdgcn_wmma_f32_16x16x32_bf16(
                        false, afr[mi].v, false, bfr[ni].v,
                        (short)0, acc[mi][ni], false, false);
        }
        __syncthreads();
    }

    // epilogue: f32 C layout (lane -> N=lane%16, M=8*(lane/16)+v)
#pragma unroll
    for (int mi = 0; mi < 2; ++mi) {
#pragma unroll
        for (int ni = 0; ni < 4; ++ni) {
            const int gm = m0 + wm + mi * 16 + lhalf * 8;
            const int gn = n0 + wn + ni * 16 + lmod;
#pragma unroll
            for (int v = 0; v < 8; ++v)
                pre[(size_t)(gm + v) * N_DIM + gn] = acc[mi][ni][v];
        }
    }
}

// ---------------------------------------------------------------------------
// Kernel 3: LayerNorm over 3072 + gate transform, IN PLACE on pre:
//   pre[c] <- a = 1-sig(yg) ; pre[c+1024] <- b = sig(yg)*yx ; pre[c+2048] <- hg
// ---------------------------------------------------------------------------
__global__ __launch_bounds__(256) void ln_gate_kernel(
    float* __restrict__ pre,
    const float* __restrict__ gamma,
    const float* __restrict__ beta)
{
    __shared__ float ssum[256];
    __shared__ float ssq[256];

    float* p = pre + (size_t)blockIdx.x * N_DIM;
    const int tid = threadIdx.x;

    float s = 0.f, q = 0.f;
    for (int i = tid; i < N_DIM; i += 256) {
        float v = p[i];
        s += v;
        q += v * v;
    }
    ssum[tid] = s;
    ssq[tid]  = q;
    __syncthreads();
    for (int off = 128; off > 0; off >>= 1) {
        if (tid < off) {
            ssum[tid] += ssum[tid + off];
            ssq[tid]  += ssq[tid + off];
        }
        __syncthreads();
    }
    const float inv_n = 1.0f / (float)N_DIM;
    const float mu    = ssum[0] * inv_n;
    const float var   = ssq[0] * inv_n - mu * mu;
    const float rstd  = rsqrtf(var + 1e-5f);

    for (int c = tid; c < D_DIM; c += 256) {
        float yg = (p[c]            - mu) * rstd * gamma[c]            + beta[c];
        float yx = (p[c + D_DIM]    - mu) * rstd * gamma[c + D_DIM]    + beta[c + D_DIM];
        float yh = (p[c + 2*D_DIM]  - mu) * rstd * gamma[c + 2*D_DIM]  + beta[c + 2*D_DIM];
        float g  = 1.0f / (1.0f + __expf(-yg));
        float hg = 1.0f / (1.0f + __expf(-yh));
        p[c]           = 1.0f - g;   // a
        p[c + D_DIM]   = g * yx;     // b
        p[c + 2*D_DIM] = hg;         // h_gate
    }
}

// ---------------------------------------------------------------------------
// Chunked bidirectional scan (3 passes). h_t = a_t*h_prev + b_t.
// Chunk aggregate: (A,B) <- (A*a_t, a_t*B + b_t) starting (1,0), so
// h_out = A*h_in + B. Forward channels c<512 scan ascending, backward c>=512
// descending. 16 chunks x 16 batch x 1024 ch = 262144 threads for passes 1/3.
// ---------------------------------------------------------------------------
__global__ __launch_bounds__(256) void scan_pass1(
    const float* __restrict__ pre, float* __restrict__ cA, float* __restrict__ cB)
{
    const int tid = blockIdx.x * 256 + threadIdx.x;  // 262144
    const int c = tid & 1023;
    const int b = (tid >> 10) & 15;
    const int q = tid >> 14;
    const bool fwd = (c < 512);
    const int t0 = q * CHUNK;

    float A = 1.f, Bv = 0.f;
    for (int i = 0; i < CHUNK; ++i) {
        const int t = fwd ? (t0 + i) : (t0 + CHUNK - 1 - i);
        const size_t rp = (size_t)(t * B_DIM + b) * N_DIM;
        if (i + 4 < CHUNK) {
            const int tp = fwd ? (t + 4) : (t - 4);
            __builtin_prefetch(pre + (size_t)(tp * B_DIM + b) * N_DIM + c,         0, 0);
            __builtin_prefetch(pre + (size_t)(tp * B_DIM + b) * N_DIM + D_DIM + c, 0, 0);
        }
        const float a  = pre[rp + c];
        const float bb = pre[rp + D_DIM + c];
        A  = A * a;
        Bv = fmaf(a, Bv, bb);
    }
    cA[tid] = A;
    cB[tid] = Bv;
}

__global__ __launch_bounds__(256) void scan_pass2(
    const float* __restrict__ cA, const float* __restrict__ cB,
    float* __restrict__ hin)
{
    const int tid = blockIdx.x * 256 + threadIdx.x;  // 16384
    const int c = tid & 1023;
    const int b = tid >> 10;
    const bool fwd = (c < 512);

    float state = 0.f;
    if (fwd) {
        for (int q = 0; q < NCHUNK; ++q) {
            const int idx = (q * B_DIM + b) * D_DIM + c;
            hin[idx] = state;
            state = fmaf(cA[idx], state, cB[idx]);
        }
    } else {
        for (int q = NCHUNK - 1; q >= 0; --q) {
            const int idx = (q * B_DIM + b) * D_DIM + c;
            hin[idx] = state;
            state = fmaf(cA[idx], state, cB[idx]);
        }
    }
}

__global__ __launch_bounds__(256) void scan_pass3(
    const float* __restrict__ pre, const float* __restrict__ x_in,
    const float* __restrict__ hin, float* __restrict__ out)
{
    const int tid = blockIdx.x * 256 + threadIdx.x;  // 262144
    const int c = tid & 1023;
    const int b = (tid >> 10) & 15;
    const int q = tid >> 14;
    const bool fwd = (c < 512);
    const int t0 = q * CHUNK;

    float h = hin[(q * B_DIM + b) * D_DIM + c];
    for (int i = 0; i < CHUNK; ++i) {
        const int t = fwd ? (t0 + i) : (t0 + CHUNK - 1 - i);
        if (i + 4 < CHUNK) {
            const int tp = fwd ? (t + 4) : (t - 4);
            const size_t pp = (size_t)(tp * B_DIM + b);
            __builtin_prefetch(pre  + pp * N_DIM + c,           0, 0);
            __builtin_prefetch(pre  + pp * N_DIM + D_DIM + c,   0, 0);
            __builtin_prefetch(pre  + pp * N_DIM + 2*D_DIM + c, 0, 0);
            __builtin_prefetch(x_in + pp * D_DIM + c,           0, 0);
        }
        const size_t rp = (size_t)(t * B_DIM + b) * N_DIM;
        const size_t ri = (size_t)(t * B_DIM + b) * D_DIM;
        const float a  = pre[rp + c];
        const float bv = pre[rp + D_DIM + c];
        const float hg = pre[rp + 2*D_DIM + c];
        const float x0 = x_in[ri + c];
        h = fmaf(a, h, bv);
        out[ri + c] = fmaf(hg, x0 - h, h);   // (1-hg)*h + x*hg
    }
}

// ---------------------------------------------------------------------------
// Launcher. Workspace layout (bytes):
//   pre  : M*N*4              = 402,653,184
//   Xb   : M*K*2              =  67,108,864
//   Wt   : N*K*2              =   6,291,456
//   cA   : 16*16*1024*4       =   1,048,576
//   cB   : 1,048,576 ; hin : 1,048,576          (total ~479 MB)
// ---------------------------------------------------------------------------
extern "C" void kernel_launch(void* const* d_in, const int* in_sizes, int n_in,
                              void* d_out, int out_size, void* d_ws, size_t ws_size,
                              hipStream_t stream)
{
    const float* x_in  = (const float*)d_in[0];
    const float* W     = (const float*)d_in[1];
    const float* gamma = (const float*)d_in[2];
    const float* beta  = (const float*)d_in[3];
    float* out = (float*)d_out;

    const size_t PRE_B = (size_t)M_DIM * N_DIM * sizeof(float);
    const size_t XB_B  = (size_t)M_DIM * K_DIM * sizeof(unsigned short);
    const size_t WT_B  = (size_t)N_DIM * K_DIM * sizeof(unsigned short);
    const size_t AGG_B = (size_t)NCHUNK * B_DIM * D_DIM * sizeof(float);

    char* ws = (char*)d_ws;
    float*          pre = (float*)ws;
    unsigned short* Xb  = (unsigned short*)(ws + PRE_B);
    unsigned short* Wt  = (unsigned short*)(ws + PRE_B + XB_B);
    float*          cA  = (float*)(ws + PRE_B + XB_B + WT_B);
    float*          cB  = (float*)(ws + PRE_B + XB_B + WT_B + AGG_B);
    float*          hin = (float*)(ws + PRE_B + XB_B + WT_B + 2 * AGG_B);

    {   // X -> bf16 ; W -> bf16 transposed
        const int nx = M_DIM * K_DIM;
        cvt_bf16_kernel<<<(nx + 255) / 256, 256, 0, stream>>>(x_in, Xb, nx);
        dim3 tg(N_DIM / 32, K_DIM / 32);   // (96, 32)
        wt_kernel<<<tg, 256, 0, stream>>>(W, Wt);
    }
    {   // GEMM -> pre
        dim3 grid(N_DIM / 128, M_DIM / 128);  // (24, 256)
        gemm_bf16_kernel<<<grid, 256, 0, stream>>>(Xb, Wt, pre);
    }
    ln_gate_kernel<<<M_DIM, 256, 0, stream>>>(pre, gamma, beta);

    scan_pass1<<<(NCHUNK * B_DIM * D_DIM) / 256, 256, 0, stream>>>(pre, cA, cB);
    scan_pass2<<<(B_DIM * D_DIM) / 256, 256, 0, stream>>>(cA, cB, hin);
    scan_pass3<<<(NCHUNK * B_DIM * D_DIM) / 256, 256, 0, stream>>>(pre, x_in, hin, out);
}
